// PreModel_6141803233546
// MI455X (gfx1250) — compile-verified, matrix operand order checked by reading, not run on hardware
//
#include <hip/hip_runtime.h>
#include <hip/hip_bf16.h>

#define NNODES 8192
#define NEDGES 262144
#define FEATD  512
#define HIDD   128

typedef __attribute__((ext_vector_type(16))) __bf16 v16bf;
typedef __attribute__((ext_vector_type(8)))  float  v8f;

// ---------------- elementwise helpers ----------------
__global__ void zero_f32_kernel(float* p, int n) {
    int i = blockIdx.x * blockDim.x + threadIdx.x;
    if (i < n) p[i] = 0.0f;
}
__global__ void zero_i32_kernel(int* p, int n) {
    int i = blockIdx.x * blockDim.x + threadIdx.x;
    if (i < n) p[i] = 0;
}
__global__ void count_deg_kernel(const int* __restrict__ dst, int* __restrict__ deg, int e) {
    int i = blockIdx.x * blockDim.x + threadIdx.x;
    if (i < e) atomicAdd(&deg[dst[i]], 1);
}
__global__ void dinv_kernel(const int* __restrict__ deg, float* __restrict__ dinv, int n) {
    int i = blockIdx.x * blockDim.x + threadIdx.x;
    if (i < n) dinv[i] = rsqrtf((float)(deg[i] + 1));   // +1 = self loop
}
__global__ void cvt_bf16_kernel(const float* __restrict__ in, __bf16* __restrict__ out, int n) {
    int i = blockIdx.x * blockDim.x + threadIdx.x;
    if (i < n) out[i] = (__bf16)in[i];
}
// Wt[n*K + k] = (bf16) W[k*Nc + n]   (W is [K,Nc] row-major)
__global__ void transpose_cvt_kernel(const float* __restrict__ W, __bf16* __restrict__ Wt,
                                     int K, int Nc) {
    int i = blockIdx.x * blockDim.x + threadIdx.x;
    if (i < K * Nc) {
        int k = i / Nc, n = i % Nc;
        Wt[(size_t)n * K + k] = (__bf16)W[i];
    }
}

// ---------------- WMMA GEMM: C[M,Nc] = A[M,K] * Bt[Nc,K]^T ----------------
// bf16 inputs, f32 accumulate. One wave computes a 16x64 output strip:
// A-fragment reused across 4 B-fragments / 4 accumulators per K-step.
// Fragment layout per ISA 16-bit A-matrix: lane holds row (lane&15);
// elems 0..7  -> K = kb + half*8 + i
// elems 8..15 -> K = kb + 16 + half*8 + (i-8)         (half = lane>>4)
__device__ __forceinline__ v16bf load_frag(const __bf16* __restrict__ rowp, int kb, int half) {
    union { uint4 u[2]; v16bf v; } t;
    t.u[0] = *reinterpret_cast<const uint4*>(rowp + kb + half * 8);
    t.u[1] = *reinterpret_cast<const uint4*>(rowp + kb + 16 + half * 8);
    return t.v;
}

__global__ __launch_bounds__(256)
void wmma_gemm_kernel(const __bf16* __restrict__ A, const __bf16* __restrict__ Bt,
                      float* __restrict__ C, int M, int K, int Nc) {
    const int lane = threadIdx.x & 31;
    const int wave = threadIdx.x >> 5;       // 8 waves -> block covers 128 rows x 64 cols
    const int half = lane >> 4;
    const int l16  = lane & 15;

    const int rowA    = blockIdx.y * 128 + wave * 16 + l16;
    const int colBase = blockIdx.x * 64;

    const __bf16* arow = A  + (size_t)rowA * K;
    const __bf16* b0r  = Bt + (size_t)(colBase +  0 + l16) * K;
    const __bf16* b1r  = Bt + (size_t)(colBase + 16 + l16) * K;
    const __bf16* b2r  = Bt + (size_t)(colBase + 32 + l16) * K;
    const __bf16* b3r  = Bt + (size_t)(colBase + 48 + l16) * K;

    v8f acc0 = {}, acc1 = {}, acc2 = {}, acc3 = {};

    for (int kb = 0; kb < K; kb += 32) {
        v16bf a  = load_frag(arow, kb, half);
        v16bf b0 = load_frag(b0r,  kb, half);
        v16bf b1 = load_frag(b1r,  kb, half);
        v16bf b2 = load_frag(b2r,  kb, half);
        v16bf b3 = load_frag(b3r,  kb, half);
        acc0 = __builtin_amdgcn_wmma_f32_16x16x32_bf16(false, a, false, b0, (short)0, acc0, false, false);
        acc1 = __builtin_amdgcn_wmma_f32_16x16x32_bf16(false, a, false, b1, (short)0, acc1, false, false);
        acc2 = __builtin_amdgcn_wmma_f32_16x16x32_bf16(false, a, false, b2, (short)0, acc2, false, false);
        acc3 = __builtin_amdgcn_wmma_f32_16x16x32_bf16(false, a, false, b3, (short)0, acc3, false, false);
    }

    // D layout: lane -> col n = lane&15 ; VGPR r -> row r + 8*(lane>>4)
    const int mBase = blockIdx.y * 128 + wave * 16 + half * 8;
    float* crow = C + (size_t)mBase * Nc + colBase + l16;
    #pragma unroll
    for (int r = 0; r < 8; ++r) {
        size_t off = (size_t)r * Nc;
        crow[off +  0] = acc0[r];
        crow[off + 16] = acc1[r];
        crow[off + 32] = acc2[r];
        crow[off + 48] = acc3[r];
    }
}

// ---------------- edge scatter:  AGG[dst] += H[src] * dinv[src]*dinv[dst] ----------------
__global__ void scatter_kernel(const int* __restrict__ edges, const float* __restrict__ dinv,
                               const float* __restrict__ H, float* __restrict__ AGG,
                               int dim, int e) {
    const int tpe = dim >> 2;                 // threads per edge (float4 each)
    int gid = blockIdx.x * blockDim.x + threadIdx.x;
    int eid = gid / tpe;
    if (eid >= e) return;
    int c = (gid % tpe) << 2;
    int s = edges[eid];
    int d = edges[e + eid];
    float nrm = dinv[s] * dinv[d];
    const float4 v = *reinterpret_cast<const float4*>(H + (size_t)s * dim + c);
    float* o = AGG + (size_t)d * dim + c;
    atomicAdd(o + 0, v.x * nrm);
    atomicAdd(o + 1, v.y * nrm);
    atomicAdd(o + 2, v.z * nrm);
    atomicAdd(o + 3, v.w * nrm);
}

// ---------------- epilogue: relu(AGG + H*dinv^2 (self loop) + bias) ----------------
__global__ void epilogue_kernel(const float* __restrict__ AGG, const float* __restrict__ H,
                                const float* __restrict__ dinv, const float* __restrict__ bias,
                                float* __restrict__ out_f32, __bf16* __restrict__ out_bf16,
                                int dim, int total) {
    int i = blockIdx.x * blockDim.x + threadIdx.x;
    if (i >= total) return;
    int node = i / dim, c = i % dim;
    float di = dinv[node];
    float v = AGG[i] + H[i] * (di * di) + bias[c];
    v = fmaxf(v, 0.0f);
    if (out_f32)  out_f32[i]  = v;
    if (out_bf16) out_bf16[i] = (__bf16)v;
}

// ---------------- host orchestration ----------------
extern "C" void kernel_launch(void* const* d_in, const int* in_sizes, int n_in,
                              void* d_out, int out_size, void* d_ws, size_t ws_size,
                              hipStream_t stream) {
    const float* x     = (const float*)d_in[0];
    const int*   edges = (const int*)  d_in[1];   // [2,E] flat: [0..E)=src, [E..2E)=dst
    const float* We1 = (const float*)d_in[2];  const float* be1 = (const float*)d_in[3];
    const float* We2 = (const float*)d_in[4];  const float* be2 = (const float*)d_in[5];
    const float* Wa1 = (const float*)d_in[6];  const float* ba1 = (const float*)d_in[7];
    const float* Wa2 = (const float*)d_in[8];  const float* ba2 = (const float*)d_in[9];
    const float* Ws1 = (const float*)d_in[10]; const float* bs1 = (const float*)d_in[11];
    float* out = (float*)d_out;                 // [N*N A_hat][N*FEAT X_hat]

    const int N = NNODES, E = NEDGES;

    // workspace carve-up (256B aligned)
    char* ws = (char*)d_ws;
    size_t off = 0;
    auto alloc = [&](size_t bytes) -> void* {
        void* p = ws + off;
        off += (bytes + 255) & ~(size_t)255;
        return p;
    };
    int*    p_deg  = (int*)   alloc((size_t)N * 4);
    float*  p_dinv = (float*) alloc((size_t)N * 4);
    __bf16* We1t = (__bf16*) alloc((size_t)HIDD  * FEATD * 2);
    __bf16* We2t = (__bf16*) alloc((size_t)HIDD  * HIDD  * 2);
    __bf16* Wa1t = (__bf16*) alloc((size_t)HIDD  * HIDD  * 2);
    __bf16* Wa2t = (__bf16*) alloc((size_t)FEATD * HIDD  * 2);
    __bf16* Ws1t = (__bf16*) alloc((size_t)HIDD  * HIDD  * 2);
    __bf16* p_xb  = (__bf16*) alloc((size_t)N * FEATD * 2);
    __bf16* p_h1b = (__bf16*) alloc((size_t)N * HIDD  * 2);
    __bf16* p_hb  = (__bf16*) alloc((size_t)N * HIDD  * 2);
    __bf16* p_ab  = (__bf16*) alloc((size_t)N * HIDD  * 2);
    __bf16* p_sb  = (__bf16*) alloc((size_t)N * HIDD  * 2);
    float*  p_H   = (float*)  alloc((size_t)N * FEATD * 4);
    float*  p_AGG = (float*)  alloc((size_t)N * FEATD * 4);

    auto blk = [](int n) { return (n + 255) / 256; };

    // degrees + dinv
    zero_i32_kernel<<<blk(N), 256, 0, stream>>>(p_deg, N);
    count_deg_kernel<<<blk(E), 256, 0, stream>>>(edges + E, p_deg, E);
    dinv_kernel<<<blk(N), 256, 0, stream>>>(p_deg, p_dinv, N);

    // weight transpose + bf16 convert, input convert
    transpose_cvt_kernel<<<blk(FEATD * HIDD), 256, 0, stream>>>(We1, We1t, FEATD, HIDD);
    transpose_cvt_kernel<<<blk(HIDD  * HIDD), 256, 0, stream>>>(We2, We2t, HIDD, HIDD);
    transpose_cvt_kernel<<<blk(HIDD  * HIDD), 256, 0, stream>>>(Wa1, Wa1t, HIDD, HIDD);
    transpose_cvt_kernel<<<blk(HIDD * FEATD), 256, 0, stream>>>(Wa2, Wa2t, HIDD, FEATD);
    transpose_cvt_kernel<<<blk(HIDD  * HIDD), 256, 0, stream>>>(Ws1, Ws1t, HIDD, HIDD);
    cvt_bf16_kernel<<<blk(N * FEATD), 256, 0, stream>>>(x, p_xb, N * FEATD);

    // one GCN layer: H = in @ W ; AGG = scatter(H) ; act = relu(AGG + self + b)
    auto run_layer = [&](const __bf16* inb, int K, const __bf16* Wt, int Cdim,
                         const float* bias, float* act_f32, __bf16* act_bf16) {
        dim3 g(Cdim / 64, N / 128);
        wmma_gemm_kernel<<<g, 256, 0, stream>>>(inb, Wt, p_H, N, K, Cdim);
        zero_f32_kernel<<<blk(N * Cdim), 256, 0, stream>>>(p_AGG, N * Cdim);
        scatter_kernel<<<blk(E * (Cdim / 4)), 256, 0, stream>>>(edges, p_dinv, p_H, p_AGG, Cdim, E);
        epilogue_kernel<<<blk(N * Cdim), 256, 0, stream>>>(p_AGG, p_H, p_dinv, bias,
                                                           act_f32, act_bf16, Cdim, N * Cdim);
    };

    // encoder
    run_layer(p_xb,  FEATD, We1t, HIDD,  be1, nullptr, p_h1b);
    run_layer(p_h1b, HIDD,  We2t, HIDD,  be2, nullptr, p_hb);
    // attribute decoder (X_hat -> d_out after A_hat block)
    run_layer(p_hb,  HIDD,  Wa1t, HIDD,  ba1, nullptr, p_ab);
    run_layer(p_ab,  HIDD,  Wa2t, FEATD, ba2, out + (size_t)N * N, nullptr);
    // structure decoder
    run_layer(p_hb,  HIDD,  Ws1t, HIDD,  bs1, nullptr, p_sb);

    // A_hat = s @ s^T : same kernel, Bt = s itself
    dim3 gA(N / 64, N / 128);
    wmma_gemm_kernel<<<gA, 256, 0, stream>>>(p_sb, p_sb, out, N, HIDD, N);
}
